// DeformableAttnDecoderLayer_24429773980415
// MI455X (gfx1250) — compile-verified, matrix operand order checked by reading, not run on hardware
//
#include <hip/hip_runtime.h>
#include <math.h>

// ---------------- types ----------------
typedef __bf16 bf16_t;
typedef bf16_t v16bf __attribute__((ext_vector_type(16)));
typedef bf16_t v8bf  __attribute__((ext_vector_type(8)));
typedef float  v8f   __attribute__((ext_vector_type(8)));

// ---------------- problem constants ----------------
#define NB     4
#define QN     1024
#define CM     256
#define HH     8
#define LL     4
#define PP     4
#define DD     32
#define DFFN   1024
#define LVTOT  21760          // 128^2+64^2+32^2+16^2
#define NQ     (NB*QN)        // 4096
#define MVAL   (NB*LVTOT)     // 87040

// Load one 16x32 bf16 WMMA fragment half for this lane.
// base points at (row_start + kbase) in a row-major [rows,K] bf16 matrix.
// ISA 7.12.2: lane holds k in [kbase, kbase+8) and [kbase+16, kbase+24).
__device__ __forceinline__ v16bf load_frag(const bf16_t* __restrict__ base) {
    v8bf lo = *(const v8bf*)(base);
    v8bf hi = *(const v8bf*)(base + 16);
    v16bf f;
#pragma unroll
    for (int i = 0; i < 8; ++i) { f[i] = lo[i]; f[i + 8] = hi[i]; }
    return f;
}

// ---------------- generic bf16 WMMA GEMM ----------------
// out[M,Nn] = A[M,K](bf16) * Wt[Nn,K](bf16, pre-transposed) + bias, opt relu,
// opt per-row zero mask, output either f32 (outF) or bf16 (outB).
// Requires: M %128==0, Nn %64==0, K %32==0.
__global__ __launch_bounds__(256) void gemm_bf16_wmma(
    const bf16_t* __restrict__ A, const bf16_t* __restrict__ Wt,
    const float* __restrict__ bias,
    float* __restrict__ outF, bf16_t* __restrict__ outB,
    const unsigned char* __restrict__ rowmask,
    int M, int K, int Nn, int relu)
{
    const int lane = threadIdx.x & 31;
    const int wave = threadIdx.x >> 5;
    const int half = lane >> 4;          // 0/1 -> k-base 0/8
    const int lm   = lane & 15;
    const int m0 = blockIdx.x * 128 + wave * 16;
    const int n0 = blockIdx.y * 64;
    if (m0 >= M) return;

    v8f acc[4] = {};

    const bf16_t* aptr = A  + (size_t)(m0 + lm) * K + half * 8;
    const bf16_t* bptr = Wt + (size_t)(n0 + lm) * K + half * 8;

    for (int k0 = 0; k0 < K; k0 += 32) {
        v16bf af = load_frag(aptr + k0);
        v16bf bfrag[4];
#pragma unroll
        for (int t = 0; t < 4; ++t)
            bfrag[t] = load_frag(bptr + (size_t)(16 * t) * K + k0);
#pragma unroll
        for (int t = 0; t < 4; ++t)
            acc[t] = __builtin_amdgcn_wmma_f32_16x16x32_bf16(
                false, af, false, bfrag[t], (short)0, acc[t], false, false);
    }

#pragma unroll
    for (int t = 0; t < 4; ++t) {
        const int gn = n0 + t * 16 + lm;
        const float bv = bias ? bias[gn] : 0.0f;
#pragma unroll
        for (int r = 0; r < 8; ++r) {
            const int gm = m0 + r + half * 8;
            float v = acc[t][r] + bv;
            if (relu) v = v > 0.0f ? v : 0.0f;
            if (rowmask && rowmask[gm]) v = 0.0f;
            const size_t oi = (size_t)gm * Nn + gn;
            if (outB) outB[oi] = (bf16_t)v;
            else      outF[oi] = v;
        }
    }
}

// ---------------- small helper kernels ----------------
// o = bf16(a + b)  (b may be null -> plain convert)
__global__ void addcvt_kernel(const float* __restrict__ a,
                              const float* __restrict__ b,
                              bf16_t* __restrict__ o, size_t n)
{
    size_t i = (size_t)blockIdx.x * blockDim.x + threadIdx.x;
    if (i >= n) return;
    float v = a[i] + (b ? b[i] : 0.0f);
    o[i] = (bf16_t)v;
}

// Wt[n*K+k] = bf16(W[k*Nn+n])   (one-time weight transpose+convert)
__global__ void transpose_cvt_kernel(const float* __restrict__ W,
                                     bf16_t* __restrict__ Wt, int K, int Nn)
{
    int i = blockIdx.x * blockDim.x + threadIdx.x;
    if (i >= K * Nn) return;
    int n = i / K, k = i - n * K;
    Wt[i] = (bf16_t)W[(size_t)k * Nn + n];
}

// in-place softmax over contiguous groups of 16
__global__ void softmax16_kernel(float* __restrict__ logits, int rows)
{
    int r = blockIdx.x * blockDim.x + threadIdx.x;
    if (r >= rows) return;
    float* p = logits + (size_t)r * 16;
    float v[16], m = -1e30f;
#pragma unroll
    for (int i = 0; i < 16; ++i) { v[i] = p[i]; m = fmaxf(m, v[i]); }
    float s = 0.0f;
#pragma unroll
    for (int i = 0; i < 16; ++i) { v[i] = expf(v[i] - m); s += v[i]; }
    float inv = 1.0f / s;
#pragma unroll
    for (int i = 0; i < 16; ++i) p[i] = v[i] * inv;
}

// Multi-scale deformable sampling. One wave per (n,q,h); lane = channel d.
// value bf16 [N, LV, H*D]; offsets f32 [N,Q,H,L,P,2]; attnw f32 [N,Q,H,L,P];
// refp f32 [N,Q,L,2]; sampled bf16 [N*Q, H*D].
__global__ __launch_bounds__(256) void ms_sample_kernel(
    const bf16_t* __restrict__ value, const float* __restrict__ offs,
    const float* __restrict__ attnw, const float* __restrict__ refp,
    bf16_t* __restrict__ sampled)
{
    const int gw   = (int)((blockIdx.x * blockDim.x + threadIdx.x) >> 5);
    const int lane = threadIdx.x & 31;
    const int h    = gw & 7;
    const int nq   = gw >> 3;                 // n*Q + q
    if (nq >= NQ) return;
    const int n = nq >> 10;

    const float* off = offs  + (size_t)nq * (HH * LL * PP * 2) + h * (LL * PP * 2);
    const float* aw  = attnw + (size_t)nq * (HH * LL * PP)     + h * (LL * PP);
    const float* rp  = refp  + (size_t)nq * (LL * 2);
    const bf16_t* vb = value + (size_t)n * LVTOT * CM + h * DD + lane;

    const int WT[4] = {128, 64, 32, 16};
    const int ST[4] = {0, 16384, 20480, 21504};

    float acc = 0.0f;
#pragma unroll
    for (int l = 0; l < 4; ++l) {
        const int wl = WT[l], hl = WT[l], st = ST[l];
        // x = ref_x*w + off_x - 0.5 (normalizer folds out), same for y
        const float fx = rp[l * 2 + 0] * (float)wl - 0.5f;
        const float fy = rp[l * 2 + 1] * (float)hl - 0.5f;
#pragma unroll
        for (int p = 0; p < 4; ++p) {
            const float x = fx + off[(l * 4 + p) * 2 + 0];
            const float y = fy + off[(l * 4 + p) * 2 + 1];
            const float x0f = floorf(x), y0f = floorf(y);
            const float dx = x - x0f, dy = y - y0f;
            const int x0 = (int)x0f, y0 = (int)y0f;
            const float a = aw[l * 4 + p];
            const float w00 = a * (1.0f - dy) * (1.0f - dx);
            const float w01 = a * (1.0f - dy) * dx;
            const float w10 = a * dy * (1.0f - dx);
            const float w11 = a * dy * dx;
            if ((unsigned)y0 < (unsigned)hl) {
                const bf16_t* row = vb + (size_t)(st + y0 * wl) * CM;
                if ((unsigned)x0       < (unsigned)wl) acc += w00 * (float)row[(size_t)x0 * CM];
                if ((unsigned)(x0 + 1) < (unsigned)wl) acc += w01 * (float)row[(size_t)(x0 + 1) * CM];
            }
            if ((unsigned)(y0 + 1) < (unsigned)hl) {
                const bf16_t* row = vb + (size_t)(st + (y0 + 1) * wl) * CM;
                if ((unsigned)x0       < (unsigned)wl) acc += w10 * (float)row[(size_t)x0 * CM];
                if ((unsigned)(x0 + 1) < (unsigned)wl) acc += w11 * (float)row[(size_t)(x0 + 1) * CM];
            }
        }
    }
    sampled[(size_t)nq * CM + h * DD + lane] = (bf16_t)acc;
}

// y = LayerNorm(a+b)*gamma + beta ; one wave per 256-wide row.
__global__ __launch_bounds__(256) void add_ln_kernel(
    const float* __restrict__ a, const float* __restrict__ b,
    const float* __restrict__ gamma, const float* __restrict__ beta,
    float* __restrict__ outF, bf16_t* __restrict__ outB, int M)
{
    const int gw = (int)((blockIdx.x * blockDim.x + threadIdx.x) >> 5);
    if (gw >= M) return;
    const int lane = threadIdx.x & 31;
    const float* ar = a + (size_t)gw * CM;
    const float* br = b + (size_t)gw * CM;
    float x[8], s = 0.0f, ss = 0.0f;
#pragma unroll
    for (int j = 0; j < 8; ++j) {
        const int idx = lane + 32 * j;
        const float v = ar[idx] + br[idx];
        x[j] = v; s += v; ss += v * v;
    }
#pragma unroll
    for (int o = 16; o > 0; o >>= 1) {
        s  += __shfl_xor(s,  o, 32);
        ss += __shfl_xor(ss, o, 32);
    }
    const float mean = s * (1.0f / CM);
    const float var  = ss * (1.0f / CM) - mean * mean;
    const float rstd = rsqrtf(var + 1e-5f);
#pragma unroll
    for (int j = 0; j < 8; ++j) {
        const int idx = lane + 32 * j;
        const float y = (x[j] - mean) * rstd * gamma[idx] + beta[idx];
        if (outF) outF[(size_t)gw * CM + idx] = y;
        if (outB) outB[(size_t)gw * CM + idx] = (bf16_t)y;
    }
}

// ---------------- launch ----------------
extern "C" void kernel_launch(void* const* d_in, const int* in_sizes, int n_in,
                              void* d_out, int out_size, void* d_ws, size_t ws_size,
                              hipStream_t stream)
{
    (void)in_sizes; (void)n_in; (void)out_size; (void)ws_size;

    const float* tgt   = (const float*)d_in[0];
    const float* qpos  = (const float*)d_in[1];
    const float* refp  = (const float*)d_in[2];
    const float* src   = (const float*)d_in[3];
    const unsigned char* mask = (const unsigned char*)d_in[4];
    const float* W_off = (const float*)d_in[5];   const float* b_off = (const float*)d_in[6];
    const float* W_attn= (const float*)d_in[7];   const float* b_attn= (const float*)d_in[8];
    const float* W_val = (const float*)d_in[9];   const float* b_val = (const float*)d_in[10];
    const float* W_out = (const float*)d_in[11];  const float* b_out = (const float*)d_in[12];
    const float* g1    = (const float*)d_in[13];  const float* be1   = (const float*)d_in[14];
    const float* W1    = (const float*)d_in[15];  const float* b1    = (const float*)d_in[16];
    const float* W2    = (const float*)d_in[17];  const float* b2    = (const float*)d_in[18];
    const float* g3    = (const float*)d_in[19];  const float* be3   = (const float*)d_in[20];
    float* out = (float*)d_out;

    // workspace carve-up
    char* ws = (char*)d_ws;
    size_t woff = 0;
    auto alloc = [&](size_t bytes) -> char* {
        char* p = ws + woff;
        woff = (woff + bytes + 255) & ~(size_t)255;
        return p;
    };
    bf16_t* qb      = (bf16_t*)alloc((size_t)NQ * CM * 2);
    bf16_t* srcb    = (bf16_t*)alloc((size_t)MVAL * CM * 2);
    bf16_t* wofft   = (bf16_t*)alloc((size_t)CM * CM * 2);
    bf16_t* wattnt  = (bf16_t*)alloc((size_t)128 * CM * 2);
    bf16_t* wvalt   = (bf16_t*)alloc((size_t)CM * CM * 2);
    bf16_t* woutt   = (bf16_t*)alloc((size_t)CM * CM * 2);
    bf16_t* w1t     = (bf16_t*)alloc((size_t)DFFN * CM * 2);
    bf16_t* w2t     = (bf16_t*)alloc((size_t)CM * DFFN * 2);
    float*  offsets = (float*)alloc((size_t)NQ * CM * 4);
    float*  logits  = (float*)alloc((size_t)NQ * 128 * 4);
    bf16_t* valueb  = (bf16_t*)alloc((size_t)MVAL * CM * 2);
    bf16_t* sampb   = (bf16_t*)alloc((size_t)NQ * CM * 2);
    float*  attnout = (float*)alloc((size_t)NQ * CM * 4);
    float*  xbuf    = (float*)alloc((size_t)NQ * CM * 4);
    bf16_t* xb      = (bf16_t*)alloc((size_t)NQ * CM * 2);
    bf16_t* hb      = (bf16_t*)alloc((size_t)NQ * DFFN * 2);
    float*  ffn     = (float*)alloc((size_t)NQ * CM * 4);

    // 1) weights -> bf16, transposed to [N,K]
    auto tc = [&](const float* W, bf16_t* Wt, int K, int Nn) {
        int cnt = K * Nn;
        transpose_cvt_kernel<<<(cnt + 255) / 256, 256, 0, stream>>>(W, Wt, K, Nn);
    };
    tc(W_off,  wofft,  CM, CM);
    tc(W_attn, wattnt, CM, 128);
    tc(W_val,  wvalt,  CM, CM);
    tc(W_out,  woutt,  CM, CM);
    tc(W1,     w1t,    CM, DFFN);
    tc(W2,     w2t,    DFFN, CM);

    // 2) q = bf16(tgt + query_pos); srcb = bf16(src)
    {
        size_t n = (size_t)NQ * CM;
        addcvt_kernel<<<(unsigned)((n + 255) / 256), 256, 0, stream>>>(tgt, qpos, qb, n);
        size_t n2 = (size_t)MVAL * CM;
        addcvt_kernel<<<(unsigned)((n2 + 255) / 256), 256, 0, stream>>>(src, nullptr, srcb, n2);
    }

    // 3) offsets = q @ W_off + b_off         [4096 x 256]
    gemm_bf16_wmma<<<dim3(NQ / 128, CM / 64), 256, 0, stream>>>(
        qb, wofft, b_off, offsets, nullptr, nullptr, NQ, CM, CM, 0);
    // 4) logits = q @ W_attn + b_attn        [4096 x 128]
    gemm_bf16_wmma<<<dim3(NQ / 128, 128 / 64), 256, 0, stream>>>(
        qb, wattnt, b_attn, logits, nullptr, nullptr, NQ, CM, 128, 0);
    // 5) softmax over (L*P)=16 groups
    softmax16_kernel<<<(NQ * HH + 255) / 256, 256, 0, stream>>>(logits, NQ * HH);
    // 6) value = (src @ W_val + b_val) masked -> bf16  [87040 x 256]
    gemm_bf16_wmma<<<dim3(MVAL / 128, CM / 64), 256, 0, stream>>>(
        srcb, wvalt, b_val, nullptr, valueb, mask, MVAL, CM, CM, 0);
    // 7) deformable sampling -> sampb bf16 [4096 x 256]
    {
        int waves = NQ * HH;                    // 32768 waves, 8 per block
        ms_sample_kernel<<<waves / 8, 256, 0, stream>>>(valueb, offsets, logits, refp, sampb);
    }
    // 8) attn_out = sampled @ W_out + b_out  (f32)
    gemm_bf16_wmma<<<dim3(NQ / 128, CM / 64), 256, 0, stream>>>(
        sampb, woutt, b_out, attnout, nullptr, nullptr, NQ, CM, CM, 0);
    // 9) x = LN(tgt + attn_out)  -> f32 + bf16
    add_ln_kernel<<<(NQ * 32) / 256, 256, 0, stream>>>(tgt, attnout, g1, be1, xbuf, xb, NQ);
    // 10) h = relu(x @ W1 + b1) -> bf16 [4096 x 1024]
    gemm_bf16_wmma<<<dim3(NQ / 128, DFFN / 64), 256, 0, stream>>>(
        xb, w1t, b1, nullptr, hb, nullptr, NQ, CM, DFFN, 1);
    // 11) ffn = h @ W2 + b2 (f32)
    gemm_bf16_wmma<<<dim3(NQ / 128, CM / 64), 256, 0, stream>>>(
        hb, w2t, b2, ffn, nullptr, nullptr, NQ, DFFN, CM, 0);
    // 12) out = LN(x + ffn)
    add_ln_kernel<<<(NQ * 32) / 256, 256, 0, stream>>>(xbuf, ffn, g3, be3, out, nullptr, NQ);
}
